// AlignmentOne_42563125903563
// MI455X (gfx1250) — matrix operand order repaired and checked by reading, not compiled, not run on hardware
//
#include <hip/hip_runtime.h>
#include <stdint.h>

// ---------------- CDNA5 WMMA types ----------------
typedef _Float16 v16h __attribute__((ext_vector_type(16)));
typedef _Float16 h8   __attribute__((ext_vector_type(8)));
typedef _Float16 h4   __attribute__((ext_vector_type(4)));
typedef float    v8f  __attribute__((ext_vector_type(8)));

#define B_    32
#define L_    1024
#define H_    768
#define BM_   64            // output rows per workgroup
#define NW_   4             // waves per workgroup
#define BLK_  128           // threads per workgroup
#define LC_   32            // l (key) chunk = WMMA K for GEMM2
#define NLC_  (L_ / LC_)    // 32 chunks

// LDS strides (in halves / floats), padded for conflict-free fragment reads
#define QSTR_  (H_ + 8)     // 776  -> row stride 1552B, %256dw = 4 banks/lane
#define KSTR_  (H_ + 16)    // 784
#define PSTR_  (L_ + 8)     // 1032
#define VTSTR_ (L_ + 16)    // 1040
#define CMSTR_ 33

#define Q_OFF   0
#define Q_BYTES (BM_ * QSTR_ * 2)        // 99328
#define P_OFF   (Q_OFF + Q_BYTES)
#define P_BYTES (BM_ * PSTR_ * 2)        // 132096
#define K_OFF   (P_OFF + P_BYTES)
#define K_BYTES (LC_ * KSTR_ * 2)        // 50176 (VT strip 16*1040*2=33280 reuses it)
#define CM_OFF  (K_OFF + K_BYTES)
#define CM_BYTES (BM_ * CMSTR_ * 4)      // 8448
#define LDS_TOTAL (CM_OFF + CM_BYTES)    // 290048 bytes < 320KB

static_assert(LDS_TOTAL <= 320 * 1024, "LDS budget exceeded");
static_assert(16 * VTSTR_ * 2 <= K_BYTES, "VT strip must fit in K region");

__global__ __launch_bounds__(BLK_) void
fa_sym_softmax_kernel(const float* __restrict__ a,
                      const uint8_t* __restrict__ mask,
                      const float* __restrict__ tptr,
                      float* __restrict__ out)
{
    extern __shared__ char smem[];
    _Float16* qsh  = (_Float16*)(smem + Q_OFF);
    _Float16* psh  = (_Float16*)(smem + P_OFF);
    _Float16* ksh  = (_Float16*)(smem + K_OFF);
    _Float16* vtsh = (_Float16*)(smem + K_OFF);   // reused in phase 2
    float*    cmsh = (float*)(smem + CM_OFF);

    const int tid    = threadIdx.x;
    const int wave   = tid >> 5;
    const int lane   = tid & 31;
    const int l15    = lane & 15;
    const int hiHalf = lane >> 4;                 // 0 or 1
    const int b      = blockIdx.x / (L_ / BM_);
    const int m_base = (blockIdx.x % (L_ / BM_)) * BM_;
    const float temp = tptr[0];

    // A-operand (16-bit, 16x32): loaded via boff like B -> lane<16 halves k0..15; lane>=16 k16..31 (aoff unused)
    const int aoff = hiHalf * 8;
    // B-operand (16-bit, 32x16): lane n -> contiguous 16 halves at k-offset {0|16}
    const int boff = hiHalf * 16;

    // ---------------- stage Q (f32 -> f16) ----------------
    {
        const float4* src = (const float4*)(a + ((size_t)b * L_ + m_base) * H_);
        #pragma unroll 1
        for (int i = tid; i < BM_ * (H_ / 4); i += BLK_) {
            int r = i / (H_ / 4), c = i % (H_ / 4);
            float4 v = src[(size_t)r * (H_ / 4) + c];
            h4 hv; hv.x = (_Float16)v.x; hv.y = (_Float16)v.y;
                   hv.z = (_Float16)v.z; hv.w = (_Float16)v.w;
            *(h4*)(qsh + r * QSTR_ + c * 4) = hv;
        }
    }
    __syncthreads();

    const int  qrow = wave * 16 + l15;            // this wave's Q row (A-operand lane -> row)
    const uint8_t* mcol = mask + (size_t)b * L_ * L_ + (size_t)(m_base + wave * 16 + hiHalf * 8);

    // ============ Phase 1: P = exp(masked(Q K^T * t) - chunkmax), chunk maxima ============
    #pragma unroll 1
    for (int lc = 0; lc < NLC_; ++lc) {
        const int l0 = lc * LC_;
        if (lc) __syncthreads();                  // waves done reading previous K tile
        {   // stage K rows l0..l0+31 (f32 -> f16)
            const float4* src = (const float4*)(a + ((size_t)b * L_ + l0) * H_);
            #pragma unroll 1
            for (int i = tid; i < LC_ * (H_ / 4); i += BLK_) {
                int r = i / (H_ / 4), c = i % (H_ / 4);
                float4 v = src[(size_t)r * (H_ / 4) + c];
                h4 hv; hv.x = (_Float16)v.x; hv.y = (_Float16)v.y;
                       hv.z = (_Float16)v.z; hv.w = (_Float16)v.w;
                *(h4*)(ksh + r * KSTR_ + c * 4) = hv;
            }
        }
        __syncthreads();

        v8f acc0 = {}; v8f acc1 = {};
        #pragma unroll 2
        for (int hc = 0; hc < H_ / 32; ++hc) {
            const int h0 = hc * 32;
            union { v16h v; h8 h[2]; } af;
            af.h[0] = *(const h8*)(qsh + qrow * QSTR_ + h0 + boff);
            af.h[1] = *(const h8*)(qsh + qrow * QSTR_ + h0 + boff + 8);
            v16h b0 = *(const v16h*)(ksh + l15 * KSTR_ + h0 + boff);
            v16h b1 = *(const v16h*)(ksh + (16 + l15) * KSTR_ + h0 + boff);
            acc0 = __builtin_amdgcn_wmma_f32_16x16x32_f16(false, af.v, false, b0,
                                                          (short)0, acc0, false, false);
            acc1 = __builtin_amdgcn_wmma_f32_16x16x32_f16(false, af.v, false, b1,
                                                          (short)0, acc1, false, false);
        }

        // D layout: vgpr r, lane j -> (m = r + 8*(j>>4), n = j&15). mask index is [b, l, m].
        uint64_t mv0 = *(const uint64_t*)(mcol + (size_t)(l0 + l15) * L_);
        uint64_t mv1 = *(const uint64_t*)(mcol + (size_t)(l0 + 16 + l15) * L_);
        float s0[8], s1[8], cm[8];
        #pragma unroll
        for (int r = 0; r < 8; ++r) {
            float x0 = acc0[r] * temp, x1 = acc1[r] * temp;
            s0[r] = ((mv0 >> (8 * r)) & 0xffull) ? x0 : -1.0e7f;
            s1[r] = ((mv1 >> (8 * r)) & 0xffull) ? x1 : -1.0e7f;
            float m = fmaxf(s0[r], s1[r]);          // row max across this 32-l chunk
            m = fmaxf(m, __shfl_xor(m, 1, 32));
            m = fmaxf(m, __shfl_xor(m, 2, 32));
            m = fmaxf(m, __shfl_xor(m, 4, 32));
            m = fmaxf(m, __shfl_xor(m, 8, 32));
            cm[r] = m;
        }
        #pragma unroll
        for (int r = 0; r < 8; ++r) {
            const int prow = wave * 16 + hiHalf * 8 + r;
            psh[prow * PSTR_ + l0 + l15]      = (_Float16)__expf(s0[r] - cm[r]);
            psh[prow * PSTR_ + l0 + 16 + l15] = (_Float16)__expf(s1[r] - cm[r]);
        }
        if (l15 == 0) {
            #pragma unroll
            for (int r = 0; r < 8; ++r)
                cmsh[(wave * 16 + hiHalf * 8 + r) * CMSTR_ + lc] = cm[r];
        }
    }

    // ============ Phase 2: O = (rescaled P) * V, den via ones-WMMA, divide ============
    const int rowA = wave * 16 + l15;             // A-operand lane -> P row
    float Mrow = -3.0e38f;
    #pragma unroll 1
    for (int c = 0; c < NLC_; ++c) Mrow = fmaxf(Mrow, cmsh[rowA * CMSTR_ + c]);

    v16h vones;
    #pragma unroll
    for (int i = 0; i < 16; ++i) vones[i] = (_Float16)1.0f;

    v8f den = {}; v8f rcp = {};
    #pragma unroll 1
    for (int ht = 0; ht < H_ / 16; ++ht) {
        const int h0 = ht * 16;
        __syncthreads();                          // all waves done with ksh/vtsh
        // stage V^T strip: vtsh[hh][l] = a[b, l, h0+hh]  (f32 -> f16, transposed)
        #pragma unroll 1
        for (int l = tid; l < L_; l += BLK_) {
            const float4* src = (const float4*)(a + ((size_t)b * L_ + l) * H_ + h0);
            #pragma unroll
            for (int q = 0; q < 4; ++q) {
                float4 v = src[q];
                vtsh[(q * 4 + 0) * VTSTR_ + l] = (_Float16)v.x;
                vtsh[(q * 4 + 1) * VTSTR_ + l] = (_Float16)v.y;
                vtsh[(q * 4 + 2) * VTSTR_ + l] = (_Float16)v.z;
                vtsh[(q * 4 + 3) * VTSTR_ + l] = (_Float16)v.w;
            }
        }
        __syncthreads();

        v8f acc = {};
        #pragma unroll 1
        for (int lc = 0; lc < NLC_; ++lc) {
            const int l0 = lc * LC_;
            const float f = __expf(cmsh[rowA * CMSTR_ + lc] - Mrow);  // chunk->global rescale
            union { v16h v; h8 h[2]; } pf;
            pf.h[0] = *(const h8*)(psh + rowA * PSTR_ + l0 + boff);
            pf.h[1] = *(const h8*)(psh + rowA * PSTR_ + l0 + boff + 8);
            pf.v = pf.v * (_Float16)f;            // per-lane == per-row scale (A layout)
            v16h vf = *(const v16h*)(vtsh + l15 * VTSTR_ + l0 + boff);
            acc = __builtin_amdgcn_wmma_f32_16x16x32_f16(false, pf.v, false, vf,
                                                         (short)0, acc, false, false);
            if (ht == 0)
                den = __builtin_amdgcn_wmma_f32_16x16x32_f16(false, pf.v, false, vones,
                                                             (short)0, den, false, false);
        }
        if (ht == 0) {
            #pragma unroll
            for (int r = 0; r < 8; ++r) rcp[r] = 1.0f / den[r];
        }
        float* orow = out + ((size_t)b * L_ + m_base + wave * 16 + hiHalf * 8) * H_ + h0 + l15;
        #pragma unroll
        for (int r = 0; r < 8; ++r)
            orow[(size_t)r * H_] = acc[r] * rcp[r];
    }
}

extern "C" void kernel_launch(void* const* d_in, const int* in_sizes, int n_in,
                              void* d_out, int out_size, void* d_ws, size_t ws_size,
                              hipStream_t stream) {
    const float*   a    = (const float*)d_in[0];
    const uint8_t* mask = (const uint8_t*)d_in[1];   // jax bool = 1 byte/elem
    const float*   temp = (const float*)d_in[2];
    float*         outp = (float*)d_out;
    (void)in_sizes; (void)n_in; (void)out_size; (void)d_ws; (void)ws_size;

    hipFuncSetAttribute((const void*)fa_sym_softmax_kernel,
                        hipFuncAttributeMaxDynamicSharedMemorySize, LDS_TOTAL);
    fa_sym_softmax_kernel<<<dim3(B_ * (L_ / BM_)), dim3(BLK_), LDS_TOTAL, stream>>>(
        a, mask, temp, outp);
}
